// Dynamic1DConv_39204461478388
// MI455X (gfx1250) — matrix operand order repaired and checked by reading, not compiled, not run on hardware
//
#include <hip/hip_runtime.h>
#include <hip/hip_bf16.h>

typedef __attribute__((ext_vector_type(16))) __bf16 v16bf;
typedef __attribute__((ext_vector_type(8)))  float  v8f;

#define BN 4
#define CH 128
#define HH 192
#define WWD 192
#define HW (HH*WWD)
#define ON 128
#define KS 7

// out[p, o] = s_p * ( sum_c W2[o,c] * x[p,c] + b2[o] )
// GEMM mapping: A = W2 (M=o, K=c) bf16 hi/lo in LDS (WMMA A-layout),
//               B = x tile (K=c, N=pixel) built in registers,
//               D = 16(o) x 16(pixel) f32 accumulators.
__global__ __launch_bounds__(256, 1)
void Dynamic1DConv_kernel(const float* __restrict__ x,
                          const float* __restrict__ kw,
                          const float* __restrict__ kb,
                          float* __restrict__ out) {
  extern __shared__ __align__(16) unsigned char smem[];
  __bf16* wbf = (__bf16*)smem;            // 32768 bf16 = 65536 B : weights hi/lo, WMMA A-layout
  float*  b2  = (float*)(smem + 65536);   // 128 f32

  const int tid = threadIdx.x;

  // ---- build K-summed weights in LDS, split into bf16 hi + lo ----
  // linear idx -> (kc, ot, lane, e); A operand (this code's contiguous K order): lane holds row M = lane%16,
  // element e maps to K = e + (lane>=16 ? 16 : 0)
  #pragma unroll 4
  for (int i = 0; i < 64; ++i) {
    int idx = i * 256 + tid;            // 0..16383
    int e   = idx & 15;
    int l2  = (idx >> 4) & 31;
    int ot  = (idx >> 9) & 7;
    int kc  = idx >> 12;
    int o   = ot * 16 + (l2 & 15);
    int kk  = e + ((l2 & 16) ? 16 : 0);
    int c   = kc * 32 + kk;
    float s = 0.f;
    #pragma unroll
    for (int k = 0; k < KS; ++k) s += kw[(o * KS + k) * CH + c];
    __bf16 h = (__bf16)s;
    __bf16 l = (__bf16)(s - (float)h);
    wbf[(((kc * 8 + ot) * 2 + 0) * 32 + l2) * 16 + e] = h;
    wbf[(((kc * 8 + ot) * 2 + 1) * 32 + l2) * 16 + e] = l;
  }
  if (tid < ON) {
    float s = 0.f;
    #pragma unroll
    for (int k = 0; k < KS; ++k) s += kb[tid * KS + k];
    b2[tid] = s;
  }
  __syncthreads();

  const v16bf* Wv = (const v16bf*)wbf;

  const int wave = tid >> 5;
  const int lane = tid & 31;
  const int ln   = lane & 15;   // pixel-within-tile (B/D N index)
  const int hi16 = lane >> 4;   // which K half this lane supplies for B

  // 32 pixels per wave, 8 waves per block: 256 pixels/block, 576 blocks
  const int p0 = (blockIdx.x * 8 + wave) * 32;
  const int b  = p0 / HW;          // HW divisible by 32: tile never crosses batch
  const int pp = p0 - b * HW;

  const float* xb = x + (size_t)b * CH * HW + pp + ln;

  v8f acc[8][2] = {};              // [o-tile][pixel-tile], zero accumulators
  float ssum[2] = {0.f, 0.f};      // per-pixel channel sums (this lane's pixel ln)

  #pragma unroll
  for (int kc = 0; kc < 4; ++kc) {
    // ---- load 32-channel slab of x for both 16-pixel tiles; convert to bf16 hi/lo ----
    v16bf bh[2], bl[2];
    #pragma unroll
    for (int t = 0; t < 2; ++t) {
      const float* px = xb + t * 16 + (size_t)(kc * 32 + hi16 * 16) * HW;
      float s = 0.f;
      #pragma unroll
      for (int e = 0; e < 16; ++e) {       // B operand, contiguous K order: K = hi16*16 + e, N = ln
        float v = px[(size_t)e * HW];
        s += v;
        __bf16 h = (__bf16)v;
        bh[t][e] = h;
        bl[t][e] = (__bf16)(v - (float)h);
      }
      ssum[t] += s;
    }
    // ---- 8 o-tiles x 2 pixel-tiles x 3 split-precision products ----
    #pragma unroll
    for (int ot = 0; ot < 8; ++ot) {
      v16bf ah = Wv[((kc * 8 + ot) * 2 + 0) * 32 + lane];
      v16bf al = Wv[((kc * 8 + ot) * 2 + 1) * 32 + lane];
      #pragma unroll
      for (int t = 0; t < 2; ++t) {
        acc[ot][t] = __builtin_amdgcn_wmma_f32_16x16x32_bf16(
            false, ah, false, bh[t], (short)0, acc[ot][t], false, false);
        acc[ot][t] = __builtin_amdgcn_wmma_f32_16x16x32_bf16(
            false, al, false, bh[t], (short)0, acc[ot][t], false, false);
        acc[ot][t] = __builtin_amdgcn_wmma_f32_16x16x32_bf16(
            false, ah, false, bl[t], (short)0, acc[ot][t], false, false);
      }
    }
  }

  // combine half-wave partial channel sums (lanes l and l^16 hold disjoint channel halves)
  ssum[0] += __shfl_xor(ssum[0], 16, 32);
  ssum[1] += __shfl_xor(ssum[1], 16, 32);

  // ---- epilogue: out[b, o, pixel] = s_p * (acc + b2[o]); lanes = consecutive pixels ----
  float* ob = out + (size_t)b * ON * HW + pp + ln;
  #pragma unroll
  for (int ot = 0; ot < 8; ++ot) {
    #pragma unroll
    for (int r = 0; r < 8; ++r) {
      int o = ot * 16 + r + hi16 * 8;      // D-layout: M = r + (lane>=16 ? 8 : 0)
      float bb = b2[o];
      ob[(size_t)o * HW + 0]  = ssum[0] * (acc[ot][0][r] + bb);
      ob[(size_t)o * HW + 16] = ssum[1] * (acc[ot][1][r] + bb);
    }
  }
}

extern "C" void kernel_launch(void* const* d_in, const int* in_sizes, int n_in,
                              void* d_out, int out_size, void* d_ws, size_t ws_size,
                              hipStream_t stream) {
  // inputs: x[4,128,192,192], offsets (unused), tumor_center (unused),
  //         kernel_w[896,128], kernel_b[896]; output: f32 [4,128,192,192]
  const float* x  = (const float*)d_in[0];
  const float* kw = (const float*)d_in[3];
  const float* kb = (const float*)d_in[4];
  float* out = (float*)d_out;

  const int pixels = BN * HW;                 // 147456
  const int blocks = pixels / 256;            // 576 (256 pixels per block)
  const size_t lds = 65536 + 512;             // weights hi/lo + b2

  Dynamic1DConv_kernel<<<dim3(blocks), dim3(256), lds, stream>>>(x, kw, kb, out);
}
